// LSTM_Classifier_1_23862838297380
// MI455X (gfx1250) — matrix operand order repaired and checked by reading, not compiled, not run on hardware
//
#include <hip/hip_runtime.h>

typedef _Float16 v16h __attribute__((ext_vector_type(16)));
typedef float    v8f  __attribute__((ext_vector_type(8)));

constexpr int kH    = 14;   // hidden size
constexpr int kT    = 128;  // timesteps
constexpr int kDin  = 12;   // input features
constexpr int kRows = 16;   // batch rows per wave (one WMMA M-tile)

// fast reciprocal: single v_rcp_f32
__device__ __forceinline__ float rcp_f(float x) {
    return __builtin_amdgcn_rcpf(x);
}

#if __has_builtin(__builtin_amdgcn_tanhf)
// CDNA5 hardware tanh: single TRANS32 instruction (v_tanh_f32)
__device__ __forceinline__ float tanh_f(float x) {
    return __builtin_amdgcn_tanhf(x);
}
// sigmoid(2a) from a PRE-SCALED argument a = 0.5*x: tanh + fma only
__device__ __forceinline__ float sigmoid_pre(float a) {
    return 0.5f + 0.5f * __builtin_amdgcn_tanhf(a);
}
__device__ __forceinline__ float sigmoid_f(float x) {
    return 0.5f + 0.5f * __builtin_amdgcn_tanhf(0.5f * x);
}
#else
__device__ __forceinline__ float sigmoid_f(float x) {
    return rcp_f(1.0f + __expf(-x));
}
__device__ __forceinline__ float sigmoid_pre(float a) {
    return sigmoid_f(2.0f * a);
}
__device__ __forceinline__ float tanh_f(float x) {
    return 1.0f - 2.0f * rcp_f(__expf(2.0f * x) + 1.0f);
}
#endif

union AFrag { v16h v; unsigned u[8]; _Float16 h[16]; };

__global__ __launch_bounds__(32) void lstm_wmma_kernel(
    const float* __restrict__ x,      // (B,T,12)
    const float* __restrict__ w1,     // (1,12)
    const float* __restrict__ b1,     // (1,)
    const float* __restrict__ Wih0,   // (56,1)
    const float* __restrict__ Whh0,   // (56,14)
    const float* __restrict__ b0,     // (56,)
    const float* __restrict__ Wih1,   // (56,14)
    const float* __restrict__ Whh1,   // (56,14)
    const float* __restrict__ b_1,    // (56,)
    const float* __restrict__ w2,     // (2,14)
    const float* __restrict__ b2,     // (2,)
    float* __restrict__ out)          // (B,2)
{
    __shared__ float    zbuf[kT * kRows];    // z[t*16+m], layer-0 scalar input
    __shared__ _Float16 stage[kRows * 32];   // A staging: row m, K=0..31 (h1|h2|0)
    __shared__ float    hlast[kRows * 16];   // final h2 for the head

    const int lane = threadIdx.x;   // 0..31 (wave32)
    const int n    = lane & 15;     // WMMA column index (gate-lane / hidden index)
    const int half = lane >> 4;     // lane half selects row/K groups
    const int m0   = blockIdx.x * kRows;

    // zero A staging once (K=28..31 stay zero forever)
    for (int i = lane; i < kRows * 32; i += 32) stage[i] = (_Float16)0.0f;

    // ---- per-lane scalar constants -------------------------------------
    float w1r[kDin];
    #pragma unroll
    for (int d = 0; d < kDin; ++d) w1r[d] = w1[d];
    const float b1s = b1[0];

    const bool okn = (n < kH);
    // Gate tiles: 0=i, 1=f, 2=g, 3=o. Sigmoid tiles (0,1,3) are pre-scaled
    // by 0.5 (weights AND biases) so sigmoid needs no pre-multiply.
    float wih0g[4], b0g[4], b1g[4];
    #pragma unroll
    for (int t = 0; t < 4; ++t) {
        const float sc = (t == 2) ? 1.0f : 0.5f;
        wih0g[t] = okn ? sc * Wih0[t * kH + n] : 0.0f;  // rank-1 input weight, L0
        b0g[t]   = okn ? sc * b0[t * kH + n]   : 0.0f;
        b1g[t]   = okn ? sc * b_1[t * kH + n]  : 0.0f;
    }

    // ---- resident f16 B fragments (32x16 per gate tile) ----------------
    // B layout used here (16-bit, 8 VGPRs, contiguous K; A uses the same): VGPR j, lane l: N = l%16, K = 2j+{0,1} + 16*(l>>4)
    // B1: rows K=0..13 = W_hh0^T (layer-1 recurrent), rest 0
    //     -> layer-1 WMMA ignores the h2 half of A (enables single A-reload/step)
    // B2: rows K=0..13 = W_ih1^T, K=14..27 = W_hh1^T (layer-2 input+recurrent fused)
    // Sigmoid tiles carry a folded 0.5 scale.
    AFrag B1[4], B2[4];
    #pragma unroll
    for (int t = 0; t < 4; ++t) {
        const float sc = (t == 2) ? 1.0f : 0.5f;
        #pragma unroll
        for (int j = 0; j < 8; ++j) {
            #pragma unroll
            for (int dj = 0; dj < 2; ++dj) {
                const int K = 2 * j + dj + 16 * half;
                float v1 = 0.0f, v2 = 0.0f;
                if (okn) {
                    const int g = t * kH + n;            // gate row (padded tile t)
                    if (K < kH) {
                        v1 = sc * Whh0[g * kH + K];
                        v2 = sc * Wih1[g * kH + K];
                    } else if (K < 2 * kH) {
                        v2 = sc * Whh1[g * kH + (K - kH)];
                    }
                }
                B1[t].h[2 * j + dj] = (_Float16)v1;
                B2[t].h[2 * j + dj] = (_Float16)v2;
            }
        }
    }

    // ---- prologue: z = tanh(sigmoid(x @ w1^T + b1)) for 16 rows x 128 t --
    #pragma unroll 1
    for (int i = 0; i < (kT * kRows) / 32; ++i) {
        const int flat = i * 32 + lane;
        const int m = flat >> 7;            // 0..15
        const int t = flat & (kT - 1);
        const float* xp = x + ((size_t)(m0 + m) * kT + t) * kDin;
        const float4 a = ((const float4*)xp)[0];
        const float4 b = ((const float4*)xp)[1];
        const float4 c = ((const float4*)xp)[2];
        float s = b1s;
        s += a.x * w1r[0] + a.y * w1r[1] + a.z * w1r[2]  + a.w * w1r[3];
        s += b.x * w1r[4] + b.y * w1r[5] + b.z * w1r[6]  + b.w * w1r[7];
        s += c.x * w1r[8] + c.y * w1r[9] + c.z * w1r[10] + c.w * w1r[11];
        zbuf[t * kRows + m] = tanh_f(sigmoid_f(s));
    }
    __syncthreads();

    // ---- recurrent loop --------------------------------------------------
    AFrag A;                                  // [h1 | h2 | 0], f16
    #pragma unroll
    for (int j = 0; j < 8; ++j) A.u[j] = 0u;

    float c1[8], c2[8], h2r[8];
    #pragma unroll
    for (int v = 0; v < 8; ++v) { c1[v] = 0.0f; c2[v] = 0.0f; h2r[v] = 0.0f; }

    const unsigned* stage32 = (const unsigned*)stage;

    #pragma unroll 1
    for (int t = 0; t < kT; ++t) {
        // z for this step; accumulator row m = v + 8*half (LDS broadcast per half)
        float zt[8];
        #pragma unroll
        for (int v = 0; v < 8; ++v) zt[v] = zbuf[t * kRows + v + 8 * half];

        // ---- layer 1: acc init = z*W_ih0 + b0, + recurrent WMMA.
        // A currently holds [h1_{t-1} | h2_{t-2}]; the stale h2 half is
        // multiplied by B1's zero rows, so only h1_{t-1} contributes.
        v8f g1[4];
        #pragma unroll
        for (int gt = 0; gt < 4; ++gt) {
            v8f acc;
            #pragma unroll
            for (int v = 0; v < 8; ++v) acc[v] = zt[v] * wih0g[gt] + b0g[gt];
            acc = __builtin_amdgcn_wmma_f32_16x16x32_f16(false, A.v, false, B1[gt].v,
                                                         (short)0, acc, false, false);
            g1[gt] = acc;
        }

        float h1v[8];
        #pragma unroll
        for (int v = 0; v < 8; ++v) {
            const float ig = sigmoid_pre(g1[0][v]);   // tiles 0,1,3 pre-scaled
            const float fg = sigmoid_pre(g1[1][v]);
            const float gg = tanh_f(g1[2][v]);
            const float og = sigmoid_pre(g1[3][v]);
            c1[v]  = fg * c1[v] + ig * gg;
            h1v[v] = og * tanh_f(c1[v]);
        }

        // publish h1_t (K = n); C layout: lane n = hidden, VGPR v -> row v+8*half
        if (okn) {
            #pragma unroll
            for (int v = 0; v < 8; ++v)
                stage[(v + 8 * half) * 32 + n] = (_Float16)h1v[v];
        }
        __syncthreads();

        // single A reload per step: A = [h1_t | h2_{t-1}] (A layout: lane = row M)
        #pragma unroll
        for (int j = 0; j < 8; ++j) {
            const int p = j + 8 * half;
            A.u[j] = stage32[n * 16 + p];
        }
        __syncthreads();   // loads complete before h2_t overwrites K=14..27

        // ---- layer 2: single WMMA fuses input (K=0..13) + recurrent (K=14..27)
        v8f g2[4];
        #pragma unroll
        for (int gt = 0; gt < 4; ++gt) {
            v8f acc;
            #pragma unroll
            for (int v = 0; v < 8; ++v) acc[v] = b1g[gt];
            acc = __builtin_amdgcn_wmma_f32_16x16x32_f16(false, A.v, false, B2[gt].v,
                                                         (short)0, acc, false, false);
            g2[gt] = acc;
        }

        #pragma unroll
        for (int v = 0; v < 8; ++v) {
            const float ig = sigmoid_pre(g2[0][v]);
            const float fg = sigmoid_pre(g2[1][v]);
            const float gg = tanh_f(g2[2][v]);
            const float og = sigmoid_pre(g2[3][v]);
            c2[v]  = fg * c2[v] + ig * gg;
            h2r[v] = og * tanh_f(c2[v]);
        }

        // publish h2_t (K = 14+n); consumed by next iteration's A reload
        if (okn) {
            #pragma unroll
            for (int v = 0; v < 8; ++v)
                stage[(v + 8 * half) * 32 + kH + n] = (_Float16)h2r[v];
        }
        // next iteration's __syncthreads (after h1 store) orders this store
        // against the next A reload; within a wave DS ops are in-order anyway.
    }

    // ---- head: logits = tanh(h2 @ w2^T + b2); softmax over 2 classes ----
    __syncthreads();
    if (okn) {
        #pragma unroll
        for (int v = 0; v < 8; ++v) hlast[(v + 8 * half) * 16 + n] = h2r[v];
    }
    __syncthreads();

    if (lane < kRows) {
        const int m = lane;
        float s0 = b2[0], s1 = b2[1];
        #pragma unroll
        for (int j = 0; j < kH; ++j) {
            const float h = hlast[m * 16 + j];
            s0 += h * w2[j];
            s1 += h * w2[kH + j];
        }
        const float l0 = tanh_f(s0), l1 = tanh_f(s1);
        const float p0 = sigmoid_f(l0 - l1);       // softmax over 2 classes
        out[(size_t)(m0 + m) * 2 + 0] = p0;
        out[(size_t)(m0 + m) * 2 + 1] = 1.0f - p0;
    }
}

extern "C" void kernel_launch(void* const* d_in, const int* in_sizes, int n_in,
                              void* d_out, int out_size, void* d_ws, size_t ws_size,
                              hipStream_t stream) {
    const float* x    = (const float*)d_in[0];
    const float* w1   = (const float*)d_in[1];
    const float* b1   = (const float*)d_in[2];
    const float* Wih0 = (const float*)d_in[3];
    const float* Whh0 = (const float*)d_in[4];
    const float* b0   = (const float*)d_in[5];
    const float* Wih1 = (const float*)d_in[6];
    const float* Whh1 = (const float*)d_in[7];
    const float* b_1  = (const float*)d_in[8];
    const float* w2   = (const float*)d_in[9];
    const float* b2   = (const float*)d_in[10];
    float* out = (float*)d_out;

    const int B = in_sizes[0] / (kT * kDin);     // 16384
    dim3 grid(B / kRows);                        // 1024 single-wave workgroups
    lstm_wmma_kernel<<<grid, 32, 0, stream>>>(x, w1, b1, Wih0, Whh0, b0,
                                              Wih1, Whh1, b_1, w2, b2, out);
}
